// BandMergeModule_15753940041763
// MI455X (gfx1250) — matrix operand order repaired and checked by reading, not compiled, not run on hardware
//
#include <hip/hip_runtime.h>
#include <hip/hip_bf16.h>

typedef _Float16 half_t;
typedef __attribute__((ext_vector_type(16))) _Float16 v16h;
typedef __attribute__((ext_vector_type(8)))  float    v8f;

#define NBANDS   26
#define EMBED    128
#define HIDDEN   512
#define NFRAMES  1024
#define TILE_M   128
#define LN_EPS   1e-5f

// ---------------- LDS layout (bytes), worst case (n=64, dout=128) ----------------
// Region R reused sequentially: xf32 (128x129 f32) -> h1 (128x520 f16) -> h3 (128x138 f32)
#define OFF_R     0
#define OFF_A     133120   // A-tile f16 128x136
#define OFF_H2    167936   // h2 f16 128x136
#define OFF_PB0   202752   // B panel 0: 512x24 f16
#define OFF_PB1   227328   // B panel 1: 512x24 f16
#define OFF_RED   251904   // 512 f32 partial sums
#define OFF_MU    253952   // 128 mu + 128 rstd
#define SMEM_BYTES 254976

#define XF32_STRIDE 129
#define A_STRIDE    136
#define H1_STRIDE   520
#define H2_STRIDE   136
#define PB_STRIDE   24

struct PtrPack { const float* p[1 + NBANDS * 10]; };

// f16 weight workspace: per band [W1 | W2 | Wg | Wgate], element offsets:
__host__ __device__ inline size_t band_off_elems(int band) {
    if (band <= 8)  return (size_t)band * 83968;                    // nb=16: 65536+16384+1024+1024
    if (band <= 16) return 671744 + (size_t)(band - 8) * 106496;    // nb=32: 65536+32768+4096+4096
    return 1523712 + (size_t)(band - 16) * 163840;                  // nb=64: 65536+65536+16384+16384
}
#define WS_NEED_BYTES 6324224   // band_off_elems(26) * 2

// ---- CDNA5 hardware transcendental: V_TANH_F32 (TRANS op; v_nop covers result hazard) ----
__device__ __forceinline__ float htanh(float x) {
    float y;
    asm("v_tanh_f32 %0, %1\n\tv_nop" : "=v"(y) : "v"(x));
    return y;
}
__device__ __forceinline__ float hsigmoid(float x) {   // sigmoid(x) = 0.5 + 0.5*tanh(x/2)
    return 0.5f + 0.5f * htanh(0.5f * x);
}

// ---- async copy of 16 bytes of f16 weights straight into LDS (ASYNCcnt path) ----
__device__ __forceinline__ void async_cp16(int lds_byte_off, int glob_byte_off,
                                           const half_t* uniform_base) {
    asm volatile("global_load_async_to_lds_b128 %0, %1, %2"
                 :: "v"(lds_byte_off), "v"(glob_byte_off), "s"(uniform_base)
                 : "memory");
}
__device__ __forceinline__ void wait_async0() {
    asm volatile("s_wait_asynccnt 0x0" ::: "memory");
}

// ---- fragment loaders (per CDNA5 ISA 7.12.2 wave32 layouts) ----
__device__ __forceinline__ v16h ld16(const half_t* p0, const half_t* p1) {
    v16h r;
    ((uint4*)&r)[0] = *(const uint4*)p0;   // elements 0..7
    ((uint4*)&r)[1] = *(const uint4*)p1;   // elements 8..15
    return r;
}
// A 16x32 f16: lane<16 -> M=lane, K = {k0..k0+7} U {k0+16..k0+23};
//              lane>=16 -> M=lane-16, K = {k0+8..k0+15} U {k0+24..k0+31}
__device__ __forceinline__ v16h ld_frag_a(const half_t* base, int stride, int m0, int k0, int lane) {
    const half_t* p = base + (m0 + (lane & 15)) * stride + k0 + ((lane >> 4) << 3);
    return ld16(p, p + 16);
}
// B 32x16 f16: lane l holds column N = l%16; K = k0 + {8h..8h+7} U {16+8h..16+8h+7}, h = l/16
__device__ __forceinline__ v16h ld_frag_b(const half_t* panel, int k0, int lane) {
    const half_t* p = panel + k0 * PB_STRIDE + (lane & 15);
    const int hk = (lane >> 4) << 3;
    v16h r;
    #pragma unroll
    for (int s = 0; s < 16; ++s) {
        int k = (s < 8) ? (hk + s) : (16 + hk + (s - 8));
        r[s] = p[k * PB_STRIDE];
    }
    return r;
}
__device__ __forceinline__ v8f wmma16(v16h a, v16h b, v8f c) {
    return __builtin_amdgcn_wmma_f32_16x16x32_f16(false, a, false, b, (short)0, c, false, false);
}

// =============== pre-pass: convert all weights to f16 in workspace ===============
__global__ __launch_bounds__(256) void convert_weights_kernel(PtrPack P, half_t* __restrict__ wsw) {
    int band = blockIdx.x >> 2;
    int mat  = blockIdx.x & 3;
    int nb   = band < 8 ? 16 : (band < 16 ? 32 : 64);
    int dout = 2 * nb;
    const float* src;
    size_t n, off;
    if (mat == 0)      { src = P.p[1 + band*10 + 2]; n = EMBED * HIDDEN; off = 0; }
    else if (mat == 1) { src = P.p[1 + band*10 + 4]; n = (size_t)HIDDEN * dout; off = EMBED * HIDDEN; }
    else if (mat == 2) { src = P.p[1 + band*10 + 6]; n = (size_t)dout * dout;
                         off = EMBED * HIDDEN + (size_t)HIDDEN * dout; }
    else               { src = P.p[1 + band*10 + 8]; n = (size_t)dout * dout;
                         off = EMBED * HIDDEN + (size_t)HIDDEN * dout + (size_t)dout * dout; }
    half_t* dst = wsw + band_off_elems(band) + off;
    for (size_t i = threadIdx.x; i < n; i += 256) dst[i] = (half_t)src[i];
}

// =============== fused band-merge kernel ===============
template <int NB>
__global__ __launch_bounds__(256) void band_merge_kernel(PtrPack P, float* __restrict__ out,
                                                         int band_base, int bin_base,
                                                         const half_t* __restrict__ wsw,
                                                         int preconv) {
    constexpr int DOUT = 2 * NB;
    constexpr int KT3  = DOUT / 32;

    extern __shared__ char smem[];
    float*  xf32  = (float*) (smem + OFF_R);
    half_t* h1    = (half_t*)(smem + OFF_R);
    float*  h3    = (float*) (smem + OFF_R);
    half_t* atile = (half_t*)(smem + OFF_A);
    half_t* h2h   = (half_t*)(smem + OFF_H2);
    half_t* pb0   = (half_t*)(smem + OFF_PB0);
    half_t* pb1   = (half_t*)(smem + OFF_PB1);
    float*  red   = (float*) (smem + OFF_RED);
    float*  mu_s  = (float*) (smem + OFF_MU);

    const int tid   = threadIdx.x;
    const int lane  = tid & 31;
    const int wave  = tid >> 5;
    const int bloc  = blockIdx.x >> 6;
    const int band  = band_base + bloc;
    const int tile  = blockIdx.x & 63;
    const int b_idx = tile >> 3;
    const int t0    = (tile & 7) * TILE_M;
    const int bin_off = bin_base + bloc * NB;

    const float* xg  = P.p[0];
    const float* gam = P.p[1 + band * 10 + 0];
    const float* bet = P.p[1 + band * 10 + 1];
    const float* W1  = P.p[1 + band * 10 + 2];
    const float* b1  = P.p[1 + band * 10 + 3];
    const float* W2  = P.p[1 + band * 10 + 4];
    const float* b2  = P.p[1 + band * 10 + 5];
    const float* Wg  = P.p[1 + band * 10 + 6];
    const float* bgv = P.p[1 + band * 10 + 7];
    const float* Wt  = P.p[1 + band * 10 + 8];
    const float* btv = P.p[1 + band * 10 + 9];

    const half_t* W1h = wsw + band_off_elems(band);
    const half_t* W2h = W1h + EMBED * HIDDEN;
    const half_t* Wgh = W2h + (size_t)HIDDEN * DOUT;
    const half_t* Wth = Wgh + (size_t)DOUT * DOUT;

    // ---------- Phase 0: stage x tile [t][e] f32 (coalesced over t) ----------
    #pragma unroll 4
    for (int it = 0; it < (EMBED * TILE_M) / 256; ++it) {
        int idx = it * 256 + tid;
        int e = idx >> 7;
        int t = idx & 127;
        float v = xg[(((size_t)b_idx * EMBED + e) * NBANDS + band) * NFRAMES + t0 + t];
        xf32[t * XF32_STRIDE + e] = v;
    }
    __syncthreads();

    // ---------- LayerNorm stats ----------
    {
        int t = tid & 127, h = tid >> 7;
        float s = 0.f, sq = 0.f;
        const float* row = xf32 + t * XF32_STRIDE + h * 64;
        #pragma unroll 8
        for (int e = 0; e < 64; ++e) { float v = row[e]; s += v; sq += v * v; }
        red[tid] = s; red[256 + tid] = sq;
    }
    __syncthreads();
    if (tid < 128) {
        float s  = red[tid] + red[tid + 128];
        float sq = red[256 + tid] + red[256 + tid + 128];
        float mu  = s * (1.0f / EMBED);
        float var = sq * (1.0f / EMBED) - mu * mu;
        mu_s[tid]       = mu;
        mu_s[128 + tid] = rsqrtf(var + LN_EPS);
    }
    __syncthreads();

    // ---------- normalize -> f16 A-tile [t][e] ----------
    #pragma unroll 4
    for (int it = 0; it < (EMBED * TILE_M) / 256; ++it) {
        int idx = it * 256 + tid;
        int e = idx >> 7;
        int t = idx & 127;
        float v = xf32[t * XF32_STRIDE + e];
        float a = (v - mu_s[t]) * mu_s[128 + t] * gam[e] + bet[e];
        atile[t * A_STRIDE + e] = (half_t)a;
    }
    __syncthreads();   // xf32 dead; region R becomes h1

    const int m0   = wave * 16;
    const int colL = lane & 15;
    const int rowh = (lane >> 4) << 3;
    const int krow = tid >> 1;            // staging: 2 lanes per panel row
    const int kpart = (tid & 1);          // which 16B half of the 32B row

    // A-fragments for GEMM1 held in VGPRs (K=128 -> 4 frags)
    v16h a0 = ld_frag_a(atile, A_STRIDE, m0, 0,  lane);
    v16h a1 = ld_frag_a(atile, A_STRIDE, m0, 32, lane);
    v16h a2 = ld_frag_a(atile, A_STRIDE, m0, 64, lane);
    v16h a3 = ld_frag_a(atile, A_STRIDE, m0, 96, lane);

    // ---------- GEMM1: A(128x128) @ W1(128x512) + b1 -> tanh -> h1 f16 ----------
    for (int nt = 0; nt < HIDDEN / 16; ++nt) {
        int n0 = nt * 16;
        if (preconv) {
            // 128 rows x 16 halfs: one async b128 per lane, no VGPR round-trip
            async_cp16(OFF_PB0 + krow * 48 + kpart * 16,
                       (krow * HIDDEN + kpart * 8) * 2, W1h + n0);
            wait_async0();
        } else {
            #pragma unroll
            for (int it = 0; it < (EMBED * 16) / 256; ++it) {
                int idx = it * 256 + tid;
                int k = idx >> 4, j = idx & 15;
                pb0[k * PB_STRIDE + j] = (half_t)W1[k * HIDDEN + n0 + j];
            }
        }
        __syncthreads();
        v8f acc = {};
        acc = wmma16(a0, ld_frag_b(pb0, 0,  lane), acc);
        acc = wmma16(a1, ld_frag_b(pb0, 32, lane), acc);
        acc = wmma16(a2, ld_frag_b(pb0, 64, lane), acc);
        acc = wmma16(a3, ld_frag_b(pb0, 96, lane), acc);
        float bias = b1[n0 + colL];
        #pragma unroll
        for (int i = 0; i < 8; ++i) {
            int row = m0 + rowh + i;
            h1[row * H1_STRIDE + n0 + colL] = (half_t)htanh(acc[i] + bias);
        }
        __syncthreads();
    }

    // ---------- GEMM2: h1(128x512) @ W2(512xDOUT) + b2 -> h2 f16 (2 n-tiles/pass) ----------
    for (int ntp = 0; ntp < DOUT / 32; ++ntp) {
        int n0 = ntp * 32;
        if (preconv) {
            #pragma unroll
            for (int it = 0; it < 4; ++it) {
                int k = it * 128 + krow;
                int gb = (k * DOUT + kpart * 8) * 2;
                async_cp16(OFF_PB0 + k * 48 + kpart * 16, gb, W2h + n0);
                async_cp16(OFF_PB1 + k * 48 + kpart * 16, gb, W2h + n0 + 16);
            }
            wait_async0();
        } else {
            #pragma unroll 4
            for (int it = 0; it < (HIDDEN * 16) / 256; ++it) {
                int idx = it * 256 + tid;
                int k = idx >> 4, j = idx & 15;
                pb0[k * PB_STRIDE + j] = (half_t)W2[k * DOUT + n0 + j];
                pb1[k * PB_STRIDE + j] = (half_t)W2[k * DOUT + n0 + 16 + j];
            }
        }
        __syncthreads();
        v8f acc0 = {}, acc1 = {};
        #pragma unroll 4
        for (int kt = 0; kt < HIDDEN / 32; ++kt) {
            v16h a = ld_frag_a(h1, H1_STRIDE, m0, kt * 32, lane);   // shared by both n-tiles
            acc0 = wmma16(a, ld_frag_b(pb0, kt * 32, lane), acc0);
            acc1 = wmma16(a, ld_frag_b(pb1, kt * 32, lane), acc1);
        }
        float bias0 = b2[n0 + colL], bias1 = b2[n0 + 16 + colL];
        #pragma unroll
        for (int i = 0; i < 8; ++i) {
            int row = m0 + rowh + i;
            h2h[row * H2_STRIDE + n0 + colL]      = (half_t)(acc0[i] + bias0);
            h2h[row * H2_STRIDE + n0 + 16 + colL] = (half_t)(acc1[i] + bias1);
        }
        __syncthreads();
    }

    // ---------- GEMM3: h2 @ Wg and h2 @ Wgate, GLU -> h3 f32 (region R) ----------
    constexpr int H3S = DOUT + 10;
    v16h aa[KT3];
    #pragma unroll
    for (int k = 0; k < KT3; ++k) aa[k] = ld_frag_a(h2h, H2_STRIDE, m0, k * 32, lane);

    for (int nt = 0; nt < DOUT / 16; ++nt) {
        int n0 = nt * 16;
        if (preconv) {
            if (tid < DOUT * 2) {
                int gb = (krow * DOUT + kpart * 8) * 2;
                async_cp16(OFF_PB0 + krow * 48 + kpart * 16, gb, Wgh + n0);
                async_cp16(OFF_PB1 + krow * 48 + kpart * 16, gb, Wth + n0);
            }
            wait_async0();
        } else {
            #pragma unroll
            for (int it = 0; it < (DOUT * 16) / 256; ++it) {
                int idx = it * 256 + tid;
                int k = idx >> 4, j = idx & 15;
                pb0[k * PB_STRIDE + j] = (half_t)Wg[k * DOUT + n0 + j];
                pb1[k * PB_STRIDE + j] = (half_t)Wt[k * DOUT + n0 + j];
            }
        }
        __syncthreads();
        v8f accg = {}, accs = {};
        #pragma unroll
        for (int k = 0; k < KT3; ++k) {
            accg = wmma16(aa[k], ld_frag_b(pb0, k * 32, lane), accg);
            accs = wmma16(aa[k], ld_frag_b(pb1, k * 32, lane), accs);
        }
        float biasg = bgv[n0 + colL], biast = btv[n0 + colL];
        #pragma unroll
        for (int i = 0; i < 8; ++i) {
            int row = m0 + rowh + i;
            float g = accg[i] + biasg;
            h3[row * H3S + n0 + colL] = g * hsigmoid(accs[i] + biast);
        }
        __syncthreads();
    }

    // ---------- store (transpose): h3[t][2j+c] -> out[b][bin_off+j][t0+t][c] ----------
    float2* out2 = (float2*)out;
    #pragma unroll 4
    for (int it = 0; it < (NB * TILE_M) / 256; ++it) {
        int idx = it * 256 + tid;
        int t = idx & 127;     // consecutive lanes -> consecutive t -> coalesced
        int j = idx >> 7;
        float2 v;
        v.x = h3[t * H3S + 2 * j];
        v.y = h3[t * H3S + 2 * j + 1];
        out2[(size_t)(b_idx * 1024 + bin_off + j) * NFRAMES + t0 + t] = v;
    }
}

extern "C" void kernel_launch(void* const* d_in, const int* in_sizes, int n_in,
                              void* d_out, int out_size, void* d_ws, size_t ws_size,
                              hipStream_t stream) {
    (void)in_sizes; (void)out_size;
    PtrPack P;
    int np = 1 + NBANDS * 10;
    for (int i = 0; i < np; ++i) P.p[i] = (i < n_in) ? (const float*)d_in[i] : nullptr;
    float* out = (float*)d_out;
    half_t* wsw = (half_t*)d_ws;
    int preconv = (d_ws != nullptr && ws_size >= (size_t)WS_NEED_BYTES) ? 1 : 0;

    if (preconv) {
        hipLaunchKernelGGL(convert_weights_kernel, dim3(NBANDS * 4), dim3(256), 0, stream, P, wsw);
    }
    // bands 0..7: n=16, bins [0,128); 8..15: n=32, bins [128,384); 16..25: n=64, bins [384,1024)
    hipLaunchKernelGGL((band_merge_kernel<16>), dim3(8 * 64),  dim3(256), SMEM_BYTES, stream,
                       P, out, 0, 0, wsw, preconv);
    hipLaunchKernelGGL((band_merge_kernel<32>), dim3(8 * 64),  dim3(256), SMEM_BYTES, stream,
                       P, out, 8, 128, wsw, preconv);
    hipLaunchKernelGGL((band_merge_kernel<64>), dim3(10 * 64), dim3(256), SMEM_BYTES, stream,
                       P, out, 16, 384, wsw, preconv);
}
